// MACE_51771535786395
// MI455X (gfx1250) — compile-verified, matrix operand order checked by reading, not run on hardware
//
#include <hip/hip_runtime.h>
#include <hip/hip_fp16.h>

typedef _Float16 v16h __attribute__((ext_vector_type(16)));
typedef float    v8f  __attribute__((ext_vector_type(8)));

#define NN 10000
#define EE 320000
#define KK 10
#define CC 32
#define NB 8
#define GG 8
#define RMAXf 5.0f
#define AVGf 32.0f

__device__ __constant__ int c_LM[16] = {0,1,1,1,2,2,2,2,2,3,3,3,3,3,3,3};

// fast sigmoid via hardware v_rcp_f32 (no Newton-Raphson refinement)
__device__ __forceinline__ float sigf(float x){
  return __builtin_amdgcn_rcpf(1.f + __expf(-x));
}
__device__ __forceinline__ float siluf(float x){ return x*sigf(x); }
__device__ __forceinline__ float dsiluf(float x){ float s = sigf(x); return s*(1.f + x*(1.f-s)); }

// WMMA 16-bit operand fragment coordinates:
//   element j of lane holds (idx = lane&15, k32 = 16*(j>>3) + 8*(lane>>4) + (j&7))
#define FRAG_LANE(m,k) (((m)&15) | ((((k)>>3)&1)<<4))
#define FRAG_J(k)      (((k)&7)  | ((((k)>>4)&1)<<3))
// aligned 32B vector load from a shared _Float16 array (offset must be multiple of 16)
#define LDV(arr, off)  (*(const v16h*)&(arr)[(off)])

__device__ __forceinline__ v8f wmma16(const v16h& a, const v16h& b, const v8f& c){
  return __builtin_amdgcn_wmma_f32_16x16x32_f16(false, a, false, b, (short)0, c, false, false);
}

template<class F>
__device__ __forceinline__ void store_tile(const v8f& c, int lane, F f){
#pragma unroll
  for (int v=0; v<8; ++v) f(v + (((lane>>4)&1)<<3), lane&15, c[v]);
}

// Layer-1 A operand straight from global rad (rows e0..e0+15, K=8 padded to 32)
__device__ __forceinline__ v16h fragA_rad(const float* __restrict__ rad, int e0, int lane){
  v16h d;
#pragma unroll
  for (int j=0;j<16;++j) d[j] = (_Float16)0.f;
  if (lane < 16){
    const float* rp = rad + (size_t)(e0+lane)*NB;
#pragma unroll
    for (int j=0;j<8;++j) d[j] = (_Float16)rp[j];
  }
  return d;
}

__device__ __forceinline__ v16h silu_frag(v16h z){
  v16h r;
#pragma unroll
  for (int j=0;j<16;++j) r[j] = (_Float16)siluf((float)z[j]);
  return r;
}

// Pre-swizzle a weight block into WMMA-B fragment layout (forward orientation):
// tile (kc,t): B(k32, n) = W[(kc*32+k32)*rowstride + (t*16+n)*colstride], zero for kg>=KLIM
#define SWZ_F(base, W, rowstride, colstride, NKC, NT, KLIM)                        \
  for (int i_ = tid; i_ < (NKC)*(NT)*512; i_ += 64){                               \
    int j_ = i_ & 15, l_ = (i_ >> 4) & 31, tl_ = i_ >> 9;                          \
    int kc_ = tl_ / (NT), t_ = tl_ - kc_*(NT);                                     \
    int k32_ = ((j_>>3)<<4) + (((l_>>4)&1)<<3) + (j_&7);                           \
    int kg_ = kc_*32 + k32_, n_ = l_ & 15;                                         \
    float v_ = (kg_ < (KLIM)) ? (W)[kg_*(rowstride) + (t_*16+n_)*(colstride)] : 0.f;\
    sW[((base)+tl_)*512 + l_*16 + j_] = (_Float16)v_;                              \
  }
// Transposed orientation (for g @ W^T): B(k32, n) = W[(t*16+n)*rowstride + kg*colstride]
#define SWZ_T(base, W, rowstride, colstride, NKC, NT, ROWLIM)                      \
  for (int i_ = tid; i_ < (NKC)*(NT)*512; i_ += 64){                               \
    int j_ = i_ & 15, l_ = (i_ >> 4) & 31, tl_ = i_ >> 9;                          \
    int kc_ = tl_ / (NT), t_ = tl_ - kc_*(NT);                                     \
    int k32_ = ((j_>>3)<<4) + (((l_>>4)&1)<<3) + (j_&7);                           \
    int kg_ = kc_*32 + k32_, n_ = l_ & 15;                                         \
    int rw_ = t_*16 + n_;                                                          \
    float v_ = (rw_ < (ROWLIM)) ? (W)[rw_*(rowstride) + kg_*(colstride)] : 0.f;    \
    sW[((base)+tl_)*512 + l_*16 + j_] = (_Float16)v_;                              \
  }
#define WT(tile) ((tile)*512 + lane*16)

// =========================== utility kernels ===========================
__global__ void k_zero(float* __restrict__ p, int n){
  int i = blockIdx.x*blockDim.x + threadIdx.x;
  if (i < n) p[i] = 0.f;
}

// =========================== edge geometry =============================
__global__ void k_edge_geom(const float* __restrict__ pos, const float* __restrict__ shifts,
                            const int* __restrict__ eidx,
                            float* __restrict__ u, float* __restrict__ rr,
                            float* __restrict__ sph, float* __restrict__ rad){
  int e = blockIdx.x*blockDim.x + threadIdx.x;
  if (e >= EE) return;
  int s = eidx[e], d = eidx[EE+e];
  float vx = pos[d*3+0]-pos[s*3+0]+shifts[e*3+0];
  float vy = pos[d*3+1]-pos[s*3+1]+shifts[e*3+1];
  float vz = pos[d*3+2]-pos[s*3+2]+shifts[e*3+2];
  float r = sqrtf(vx*vx+vy*vy+vz*vz + 1e-12f);
  float inv = 1.f/r;
  float x=vx*inv, y=vy*inv, z=vz*inv;
  u[e*3+0]=x; u[e*3+1]=y; u[e*3+2]=z; rr[e]=r;
  const float s3=1.73205081f, s5=2.23606798f, s15=3.87298335f, s105=10.24695077f;
  const float s35_8=2.09165007f, s21_8=1.62018517f, s7=2.64575131f;
  float* sp = &sph[e*16];
  sp[0]=1.f; sp[1]=s3*x; sp[2]=s3*y; sp[3]=s3*z;
  sp[4]=s15*x*y; sp[5]=s15*y*z; sp[6]=0.5f*s5*(3.f*z*z-1.f); sp[7]=s15*x*z;
  sp[8]=0.5f*s15*(x*x-y*y);
  sp[9]=s35_8*y*(3.f*x*x-y*y); sp[10]=s105*x*y*z; sp[11]=s21_8*y*(5.f*z*z-1.f);
  sp[12]=0.5f*s7*z*(5.f*z*z-3.f); sp[13]=s21_8*x*(5.f*z*z-1.f);
  sp[14]=0.5f*s105*z*(x*x-y*y); sp[15]=s35_8*x*(x*x-3.f*y*y);
  float t = r/RMAXf;
  float env = 0.f;
  if (r < RMAXf){
    float t2=t*t, t4=t2*t2, t5=t4*t;
    env = 1.f - 21.f*t5 + 35.f*t5*t - 15.f*t5*t2;
  }
  const float A0=0.63245553f, w=0.62831853f;
#pragma unroll
  for (int b=1;b<=NB;++b)
    rad[e*NB + b-1] = A0*__sinf(w*(float)b*r)*inv * env;
}

// =========================== node embed ================================
__global__ void k_node_embed(const float* __restrict__ na, const float* __restrict__ Wemb,
                             const float* __restrict__ Wup1, int* __restrict__ nodek,
                             float* __restrict__ scal1){
  int n = blockIdx.x; int c = threadIdx.x;
  int kk = 0;
#pragma unroll
  for (int j=0;j<KK;++j) if (na[n*KK+j] > 0.5f) kk = j;
  if (c==0) nodek[n]=kk;
  float acc = 0.f;
  for (int cp=0;cp<CC;++cp) acc += Wemb[kk*CC+cp]*Wup1[cp*CC+c];
  scal1[n*CC+c]=acc;
}

// =========================== MLP1 forward ==============================
// tiles: Wa[0..3] Wb[4..11] Wc[12..19] Wd0[20..23]
__global__ __launch_bounds__(64) void k_mlp1_fwd(const float* __restrict__ rad,
    const float* __restrict__ M1a, const float* __restrict__ M1b,
    const float* __restrict__ M1c, const float* __restrict__ M1d,
    const int* __restrict__ eidx, const float* __restrict__ scal1,
    float* __restrict__ msg1){
  __shared__ alignas(32) _Float16 sW[24*512];
  __shared__ alignas(32) _Float16 sZf[2*2*512];
  __shared__ alignas(32) _Float16 sOut[2*512];
  int tid = threadIdx.x, lane = tid&31, wv = tid>>5;
  SWZ_F(0,  M1a, 64, 1, 1, 4, 8);
  SWZ_F(4,  M1b, 64, 1, 2, 4, 64);
  SWZ_F(12, M1c, 64, 1, 2, 4, 64);
  SWZ_F(20, M1d, 128, 4, 2, 2, 64);
  __syncthreads();
  int e0 = (blockIdx.x*2 + wv)*16;
  __builtin_prefetch(&rad[(size_t)(e0+32)*NB], 0, 1);
  int zb = wv*2*512;
  // L1
  {
    v16h aR = fragA_rad(rad, e0, lane);
#pragma unroll
    for (int t=0;t<4;++t){
      v8f acc = {};
      acc = wmma16(aR, LDV(sW, WT(0+t)), acc);
      store_tile(acc, lane, [&](int r,int c0,float vv){
        int cg=t*16+c0, k=cg&31;
        sZf[zb + (cg>>5)*512 + FRAG_LANE(r,k)*16 + FRAG_J(k)] = (_Float16)siluf(vv); });
    }
  }
  // L2
  {
    v16h A0 = LDV(sZf, zb), A1 = LDV(sZf, zb+512);
#pragma unroll
    for (int t=0;t<4;++t){
      v8f acc = {};
      acc = wmma16(A0, LDV(sW, WT(4+t)), acc);
      acc = wmma16(A1, LDV(sW, WT(8+t)), acc);
      store_tile(acc, lane, [&](int r,int c0,float vv){
        int cg=t*16+c0, k=cg&31;
        sZf[zb + (cg>>5)*512 + FRAG_LANE(r,k)*16 + FRAG_J(k)] = (_Float16)siluf(vv); });
    }
  }
  // L3
  {
    v16h A0 = LDV(sZf, zb), A1 = LDV(sZf, zb+512);
#pragma unroll
    for (int t=0;t<4;++t){
      v8f acc = {};
      acc = wmma16(A0, LDV(sW, WT(12+t)), acc);
      acc = wmma16(A1, LDV(sW, WT(16+t)), acc);
      store_tile(acc, lane, [&](int r,int c0,float vv){
        int cg=t*16+c0, k=cg&31;
        sZf[zb + (cg>>5)*512 + FRAG_LANE(r,k)*16 + FRAG_J(k)] = (_Float16)siluf(vv); });
    }
  }
  // L4: 64 -> 32 (t0 slice of M1d)
  {
    v16h A0 = LDV(sZf, zb), A1 = LDV(sZf, zb+512);
#pragma unroll
    for (int t=0;t<2;++t){
      v8f acc = {};
      acc = wmma16(A0, LDV(sW, WT(20+t)), acc);
      acc = wmma16(A1, LDV(sW, WT(22+t)), acc);
      store_tile(acc, lane, [&](int r,int c0,float vv){
        sOut[wv*512 + r*32 + t*16 + c0] = (_Float16)vv; });
    }
  }
  int c = lane;
  for (int mloc=0;mloc<16;++mloc){
    int e = e0+mloc;
    int snd = eidx[e], rcv = eidx[EE+e];
    float t0 = (float)sOut[wv*512 + mloc*32 + c];
    atomicAdd(&msg1[rcv*CC+c], scal1[snd*CC+c]*t0);
  }
}

// =========================== node forward 1 ============================
__global__ void k_node_fwd1(const float* __restrict__ msg1, const float* __restrict__ Wl1,
                            const float* __restrict__ Wprod1, const float* __restrict__ Wup2,
                            const int* __restrict__ nodek,
                            float* __restrict__ A1_0, float* __restrict__ nf1_0,
                            float* __restrict__ scal2){
  int n = blockIdx.x, c = threadIdx.x;
  __shared__ float sm[CC], snf[CC];
  sm[c] = msg1[n*CC+c];
  __syncthreads();
  float acc = 0.f;
  for (int cp=0;cp<CC;++cp) acc += sm[cp]*Wl1[cp*CC + c];
  float A = acc*(1.f/AVGf);
  A1_0[n*CC+c]=A;
  int kk = nodek[n];
  float c0=Wprod1[kk*CC*3 + c*3 + 0], c1=Wprod1[kk*CC*3 + c*3 + 1], c2=Wprod1[kk*CC*3 + c*3 + 2];
  float nf = A*(c0 + c1*A + c2*A*A);
  nf1_0[n*CC+c]=nf; snf[c]=nf;
  __syncthreads();
  float s2 = 0.f;
  for (int d=0; d<CC; ++d) s2 += snf[d]*Wup2[d*CC + c];
  scal2[n*CC+c]=s2;
}

// =========================== MLP2 forward ==============================
// tiles: Wa[0..3] Wb[4..11] Wc[12..19] Wd[20..35]
__global__ __launch_bounds__(64) void k_mlp2_fwd(const float* __restrict__ rad,
    const float* __restrict__ M2a, const float* __restrict__ M2b,
    const float* __restrict__ M2c, const float* __restrict__ M2d,
    const int* __restrict__ eidx, const float* __restrict__ scal2,
    const float* __restrict__ sph,
    _Float16* __restrict__ tw2, float* __restrict__ msg2){
  __shared__ alignas(32) _Float16 sW[36*512];
  __shared__ alignas(32) _Float16 sZf[2*2*512];
  __shared__ alignas(32) _Float16 sOut[2*2048];
  int tid = threadIdx.x, lane = tid&31, wv = tid>>5;
  SWZ_F(0,  M2a, 64, 1, 1, 4, 8);
  SWZ_F(4,  M2b, 64, 1, 2, 4, 64);
  SWZ_F(12, M2c, 64, 1, 2, 4, 64);
  SWZ_F(20, M2d, 128, 1, 2, 8, 64);
  __syncthreads();
  int e0 = (blockIdx.x*2 + wv)*16;
  __builtin_prefetch(&rad[(size_t)(e0+32)*NB], 0, 1);
  int zb = wv*2*512;
  // L1..L3
  {
    v16h aR = fragA_rad(rad, e0, lane);
#pragma unroll
    for (int t=0;t<4;++t){
      v8f acc = {};
      acc = wmma16(aR, LDV(sW, WT(0+t)), acc);
      store_tile(acc, lane, [&](int r,int c0,float vv){
        int cg=t*16+c0, k=cg&31;
        sZf[zb + (cg>>5)*512 + FRAG_LANE(r,k)*16 + FRAG_J(k)] = (_Float16)siluf(vv); });
    }
  }
  {
    v16h A0 = LDV(sZf, zb), A1 = LDV(sZf, zb+512);
#pragma unroll
    for (int t=0;t<4;++t){
      v8f acc = {};
      acc = wmma16(A0, LDV(sW, WT(4+t)), acc);
      acc = wmma16(A1, LDV(sW, WT(8+t)), acc);
      store_tile(acc, lane, [&](int r,int c0,float vv){
        int cg=t*16+c0, k=cg&31;
        sZf[zb + (cg>>5)*512 + FRAG_LANE(r,k)*16 + FRAG_J(k)] = (_Float16)siluf(vv); });
    }
  }
  {
    v16h A0 = LDV(sZf, zb), A1 = LDV(sZf, zb+512);
#pragma unroll
    for (int t=0;t<4;++t){
      v8f acc = {};
      acc = wmma16(A0, LDV(sW, WT(12+t)), acc);
      acc = wmma16(A1, LDV(sW, WT(16+t)), acc);
      store_tile(acc, lane, [&](int r,int c0,float vv){
        int cg=t*16+c0, k=cg&31;
        sZf[zb + (cg>>5)*512 + FRAG_LANE(r,k)*16 + FRAG_J(k)] = (_Float16)siluf(vv); });
    }
  }
  // L4: 64 -> 128
  {
    v16h A0 = LDV(sZf, zb), A1 = LDV(sZf, zb+512);
#pragma unroll
    for (int t=0;t<8;++t){
      v8f acc = {};
      acc = wmma16(A0, LDV(sW, WT(20+t)), acc);
      acc = wmma16(A1, LDV(sW, WT(28+t)), acc);
      store_tile(acc, lane, [&](int r,int c0,float vv){
        sOut[wv*2048 + r*128 + t*16 + c0] = (_Float16)vv; });
    }
  }
  // cache tw2: 16x128 tile is contiguous in global ([e0*128, (e0+16)*128))
#pragma unroll
  for (int q=0;q<4;++q){
    int slot = lane*4 + q;
    *(v16h*)(tw2 + (size_t)e0*128 + slot*16) = LDV(sOut, wv*2048 + slot*16);
  }
  // message scatter
  for (int mloc=0;mloc<16;++mloc){
    int e = e0+mloc;
    int snd = eidx[e], rcv = eidx[EE+e];
    float sc = scal2[snd*CC+lane];
    const float* sp = &sph[e*16];
    float twv[4];
#pragma unroll
    for (int l=0;l<4;++l) twv[l] = (float)sOut[wv*2048 + mloc*128 + lane*4 + l];
#pragma unroll
    for (int mm=0;mm<16;++mm){
      int l = c_LM[mm];
      atomicAdd(&msg2[rcv*512 + lane*16 + mm], sc * twv[l] * sp[mm]);
    }
  }
}

// =========================== node forward 2 ============================
__global__ void k_node_fwd2(const float* __restrict__ msg2, const float* __restrict__ Wl2,
                            const float* __restrict__ Wprod2, const float* __restrict__ Wsc2,
                            const int* __restrict__ nodek, const float* __restrict__ nf1_0,
                            const float* __restrict__ aen, const float* __restrict__ wr1,
                            const float* __restrict__ Wr2a, const float* __restrict__ wr2b,
                            const int* __restrict__ batch,
                            float* __restrict__ A2, float* __restrict__ zbuf,
                            float* __restrict__ out){
  int n = blockIdx.x, d = threadIdx.x;
  __shared__ float sm[512], snf1[CC], snf2[CC], sz[16];
  for (int m=0;m<16;++m) sm[d*16+m] = msg2[n*512 + d*16 + m];
  snf1[d] = nf1_0[n*CC+d];
  __syncthreads();
  float pl[4] = {0.f,0.f,0.f,0.f};
#pragma unroll
  for (int m=0;m<16;++m){
    int l = c_LM[m];
    float acc = 0.f;
    for (int cc=0;cc<CC;++cc) acc += sm[cc*16+m]*Wl2[l*CC*CC + cc*CC + d];
    acc *= (1.f/AVGf);
    A2[n*512 + d*16 + m] = acc;
    pl[l] += acc*acc;
  }
  int kk = nodek[n];
  float v = 0.f;
#pragma unroll
  for (int t=0;t<4;++t) v += Wprod2[kk*CC*4 + d*4 + t]*pl[t];
  float sc = 0.f;
  for (int cc=0;cc<CC;++cc) sc += snf1[cc]*Wsc2[kk*CC*CC + cc*CC + d];
  float nf = v + sc;
  snf2[d] = nf;
  __syncthreads();
  if (d < 16){
    float zz = 0.f;
    for (int cc=0;cc<CC;++cc) zz += snf2[cc]*Wr2a[cc*16 + d];
    zbuf[n*16+d] = zz; sz[d] = zz;
  }
  __syncthreads();
  float red = snf1[d]*wr1[d] + ((d<16) ? siluf(sz[d])*wr2b[d] : 0.f);
#pragma unroll
  for (int o=16;o>=1;o>>=1) red += __shfl_xor(red, o, 32);
  if (d==0){
    float ne = aen[kk] + red;
    out[GG + n] = ne;
    atomicAdd(&out[batch[n]], ne);
  }
}

// =========================== node backward 1 ===========================
__global__ void k_node_bwd1(const float* __restrict__ zbuf, const float* __restrict__ wr2b,
                            const float* __restrict__ Wr2a, const float* __restrict__ Wsc2,
                            const float* __restrict__ Wprod2, const float* __restrict__ Wl2,
                            const int* __restrict__ nodek, const float* __restrict__ A2,
                            const float* __restrict__ wr1,
                            float* __restrict__ gnf1, float* __restrict__ gmsg2){
  int n = blockIdx.x, c = threadIdx.x;
  __shared__ float sgz[16], sg2[CC], sga[512];
  if (c < 16) sgz[c] = wr2b[c]*dsiluf(zbuf[n*16+c]);
  __syncthreads();
  float g = 0.f;
  for (int j=0;j<16;++j) g += sgz[j]*Wr2a[c*16+j];
  sg2[c] = g;
  __syncthreads();
  int kk = nodek[n];
  float gn = wr1[c];
  for (int d=0; d<CC; ++d) gn += sg2[d]*Wsc2[kk*CC*CC + c*CC + d];
  gnf1[n*CC+c] = gn;
#pragma unroll
  for (int m=0;m<16;++m){
    int l = c_LM[m];
    sga[c*16+m] = sg2[c]*Wprod2[kk*CC*4 + c*4 + l]*2.f*A2[n*512 + c*16 + m];
  }
  __syncthreads();
#pragma unroll
  for (int m=0;m<16;++m){
    int l = c_LM[m];
    float acc = 0.f;
    for (int d=0;d<CC;++d) acc += sga[d*16+m]*Wl2[l*CC*CC + c*CC + d];
    gmsg2[n*512 + c*16 + m] = acc*(1.f/AVGf);
  }
}

// =========================== MLP2 backward =============================
// phase A tiles: Wa[0..3] Wb[4..11] Wc[12..19]
// phase B tiles: MdT[0..15] McT[16..23] MbT[24..31] MaT[32..33]
__global__ __launch_bounds__(64) void k_mlp2_bwd(const float* __restrict__ rad,
    const float* __restrict__ M2a, const float* __restrict__ M2b,
    const float* __restrict__ M2c, const float* __restrict__ M2d,
    const int* __restrict__ eidx, const float* __restrict__ scal2,
    const float* __restrict__ sph, const _Float16* __restrict__ tw2,
    const float* __restrict__ gmsg2,
    float* __restrict__ gscal2, float* __restrict__ gea, float* __restrict__ grad_rad){
  __shared__ alignas(32) _Float16 sW[34*512];
  __shared__ alignas(32) _Float16 sZr[2*3*2*512];
  __shared__ alignas(32) _Float16 sG[2*4*512];
  int tid = threadIdx.x, lane = tid&31, wv = tid>>5;
  SWZ_F(0,  M2a, 64, 1, 1, 4, 8);
  SWZ_F(4,  M2b, 64, 1, 2, 4, 64);
  SWZ_F(12, M2c, 64, 1, 2, 4, 64);
  __syncthreads();
  int e0 = (blockIdx.x*2 + wv)*16;
  int zb = wv*3*2*512, gb = wv*4*512;
  // recompute, save raw pre-activations
  {
    v16h aR = fragA_rad(rad, e0, lane);
#pragma unroll
    for (int t=0;t<4;++t){
      v8f acc = {};
      acc = wmma16(aR, LDV(sW, WT(0+t)), acc);
      store_tile(acc, lane, [&](int r,int c0,float vv){
        int cg=t*16+c0, k=cg&31;
        sZr[zb + 0*1024 + (cg>>5)*512 + FRAG_LANE(r,k)*16 + FRAG_J(k)] = (_Float16)vv; });
    }
  }
  {
    v16h A0 = silu_frag(LDV(sZr, zb)), A1 = silu_frag(LDV(sZr, zb+512));
#pragma unroll
    for (int t=0;t<4;++t){
      v8f acc = {};
      acc = wmma16(A0, LDV(sW, WT(4+t)), acc);
      acc = wmma16(A1, LDV(sW, WT(8+t)), acc);
      store_tile(acc, lane, [&](int r,int c0,float vv){
        int cg=t*16+c0, k=cg&31;
        sZr[zb + 1*1024 + (cg>>5)*512 + FRAG_LANE(r,k)*16 + FRAG_J(k)] = (_Float16)vv; });
    }
  }
  {
    v16h A0 = silu_frag(LDV(sZr, zb+1024)), A1 = silu_frag(LDV(sZr, zb+1024+512));
#pragma unroll
    for (int t=0;t<4;++t){
      v8f acc = {};
      acc = wmma16(A0, LDV(sW, WT(12+t)), acc);
      acc = wmma16(A1, LDV(sW, WT(16+t)), acc);
      store_tile(acc, lane, [&](int r,int c0,float vv){
        int cg=t*16+c0, k=cg&31;
        sZr[zb + 2*1024 + (cg>>5)*512 + FRAG_LANE(r,k)*16 + FRAG_J(k)] = (_Float16)vv; });
    }
  }
  __syncthreads();
  // overlay transposed weights
  SWZ_T(0,  M2d, 128, 1, 4, 4, 64);
  SWZ_T(16, M2c, 64,  1, 2, 4, 64);
  SWZ_T(24, M2b, 64,  1, 2, 4, 64);
  SWZ_T(32, M2a, 64,  1, 2, 1, 8);
  __syncthreads();
  // per-edge: side gradients + g4 fragment build
  int c = lane;
  for (int mloc=0;mloc<16;++mloc){
    int e = e0+mloc;
    int snd = eidx[e], rcv = eidx[EE+e];
    float sc = scal2[snd*CC+c];
    const float* sp = &sph[e*16];
    float tw[4];
#pragma unroll
    for (int l=0;l<4;++l) tw[l] = (float)tw2[(size_t)e*128 + c*4 + l];
    float g2v[16];
#pragma unroll
    for (int mm=0;mm<16;++mm) g2v[mm] = gmsg2[rcv*512 + c*16 + mm];
    float gs = 0.f; float gt[4] = {0.f,0.f,0.f,0.f};
#pragma unroll
    for (int mm=0;mm<16;++mm){
      int l = c_LM[mm];
      gs    += tw[l]*sp[mm]*g2v[mm];
      gt[l] += sc*sp[mm]*g2v[mm];
    }
    atomicAdd(&gscal2[snd*CC+c], gs);
#pragma unroll
    for (int l=0;l<4;++l){
      int cg = c*4 + l, k = cg&31;
      sG[gb + (cg>>5)*512 + FRAG_LANE(mloc,k)*16 + FRAG_J(k)] = (_Float16)gt[l];
    }
#pragma unroll
    for (int mm=1;mm<16;++mm){
      float val = sc*tw[c_LM[mm]]*g2v[mm];
#pragma unroll
      for (int o=16;o>=1;o>>=1) val += __shfl_xor(val, o, 32);
      if (lane==0) gea[e*16+mm] = val;
    }
    if (lane==0) gea[e*16+0] = 0.f;
  }
  // gh3 = (g4 @ Md^T) * silu'(z3)
  {
    v16h A0 = LDV(sG, gb), A1 = LDV(sG, gb+512), A2f = LDV(sG, gb+1024), A3 = LDV(sG, gb+1536);
#pragma unroll
    for (int t=0;t<4;++t){
      v8f acc = {};
      acc = wmma16(A0,  LDV(sW, WT(0*4+t)),  acc);
      acc = wmma16(A1,  LDV(sW, WT(1*4+t)),  acc);
      acc = wmma16(A2f, LDV(sW, WT(2*4+t)),  acc);
      acc = wmma16(A3,  LDV(sW, WT(3*4+t)),  acc);
      store_tile(acc, lane, [&](int r,int c0,float vv){
        int cg=t*16+c0, k=cg&31;
        float ds = dsiluf((float)sZr[zb + 2*1024 + (cg>>5)*512 + FRAG_LANE(r,k)*16 + FRAG_J(k)]);
        sG[gb + (cg>>5)*512 + FRAG_LANE(r,k)*16 + FRAG_J(k)] = (_Float16)(vv*ds); });
    }
  }
  // gh2 = (gh3 @ Mc^T) * silu'(z2)
  {
    v16h A0 = LDV(sG, gb), A1 = LDV(sG, gb+512);
#pragma unroll
    for (int t=0;t<4;++t){
      v8f acc = {};
      acc = wmma16(A0, LDV(sW, WT(16+t)), acc);
      acc = wmma16(A1, LDV(sW, WT(20+t)), acc);
      store_tile(acc, lane, [&](int r,int c0,float vv){
        int cg=t*16+c0, k=cg&31;
        float ds = dsiluf((float)sZr[zb + 1*1024 + (cg>>5)*512 + FRAG_LANE(r,k)*16 + FRAG_J(k)]);
        sG[gb + (cg>>5)*512 + FRAG_LANE(r,k)*16 + FRAG_J(k)] = (_Float16)(vv*ds); });
    }
  }
  // gh1 = (gh2 @ Mb^T) * silu'(z1)
  {
    v16h A0 = LDV(sG, gb), A1 = LDV(sG, gb+512);
#pragma unroll
    for (int t=0;t<4;++t){
      v8f acc = {};
      acc = wmma16(A0, LDV(sW, WT(24+t)), acc);
      acc = wmma16(A1, LDV(sW, WT(28+t)), acc);
      store_tile(acc, lane, [&](int r,int c0,float vv){
        int cg=t*16+c0, k=cg&31;
        float ds = dsiluf((float)sZr[zb + 0*1024 + (cg>>5)*512 + FRAG_LANE(r,k)*16 + FRAG_J(k)]);
        sG[gb + (cg>>5)*512 + FRAG_LANE(r,k)*16 + FRAG_J(k)] = (_Float16)(vv*ds); });
    }
  }
  // grad_rad = gh1 @ Ma^T
  {
    v16h A0 = LDV(sG, gb), A1 = LDV(sG, gb+512);
    v8f acc = {};
    acc = wmma16(A0, LDV(sW, WT(32)), acc);
    acc = wmma16(A1, LDV(sW, WT(33)), acc);
    store_tile(acc, lane, [&](int r,int cc,float vv){
      if (cc < NB) grad_rad[(size_t)(e0+r)*NB + cc] = vv; });
  }
}

// =========================== node backward 2 ===========================
__global__ void k_node_bwd2(const float* __restrict__ gscal2, const float* __restrict__ Wup2,
                            const float* __restrict__ gnf1, const float* __restrict__ Wprod1,
                            const int* __restrict__ nodek, const float* __restrict__ A1_0,
                            const float* __restrict__ Wl1, float* __restrict__ gmsg1){
  int n = blockIdx.x, c = threadIdx.x;
  __shared__ float sgs[CC], sga[CC];
  sgs[c] = gscal2[n*CC+c];
  __syncthreads();
  float g = gnf1[n*CC+c];
  for (int d=0;d<CC;++d) g += sgs[d]*Wup2[c*CC+d];
  int kk = nodek[n];
  float A = A1_0[n*CC+c];
  float c0=Wprod1[kk*CC*3 + c*3 + 0], c1=Wprod1[kk*CC*3 + c*3 + 1], c2=Wprod1[kk*CC*3 + c*3 + 2];
  sga[c] = g*(c0 + 2.f*c1*A + 3.f*c2*A*A);
  __syncthreads();
  float acc = 0.f;
  for (int d=0;d<CC;++d) acc += sga[d]*Wl1[c*CC+d];
  gmsg1[n*CC+c] = acc*(1.f/AVGf);
}

// =========================== MLP1 backward =============================
// phase A tiles: Wa[0..3] Wb[4..11] Wc[12..19]
// phase B tiles: Wd0T[0..3] McT[4..11] MbT[12..19] MaT[20..21]
__global__ __launch_bounds__(64) void k_mlp1_bwd(const float* __restrict__ rad,
    const float* __restrict__ M1a, const float* __restrict__ M1b,
    const float* __restrict__ M1c, const float* __restrict__ M1d,
    const int* __restrict__ eidx, const float* __restrict__ scal1,
    const float* __restrict__ gmsg1, float* __restrict__ grad_rad){
  __shared__ alignas(32) _Float16 sW[22*512];
  __shared__ alignas(32) _Float16 sZr[2*3*2*512];
  __shared__ alignas(32) _Float16 sG[2*2*512];
  int tid = threadIdx.x, lane = tid&31, wv = tid>>5;
  SWZ_F(0,  M1a, 64, 1, 1, 4, 8);
  SWZ_F(4,  M1b, 64, 1, 2, 4, 64);
  SWZ_F(12, M1c, 64, 1, 2, 4, 64);
  __syncthreads();
  int e0 = (blockIdx.x*2 + wv)*16;
  int zb = wv*3*2*512, gb = wv*2*512;
  {
    v16h aR = fragA_rad(rad, e0, lane);
#pragma unroll
    for (int t=0;t<4;++t){
      v8f acc = {};
      acc = wmma16(aR, LDV(sW, WT(0+t)), acc);
      store_tile(acc, lane, [&](int r,int c0,float vv){
        int cg=t*16+c0, k=cg&31;
        sZr[zb + 0*1024 + (cg>>5)*512 + FRAG_LANE(r,k)*16 + FRAG_J(k)] = (_Float16)vv; });
    }
  }
  {
    v16h A0 = silu_frag(LDV(sZr, zb)), A1 = silu_frag(LDV(sZr, zb+512));
#pragma unroll
    for (int t=0;t<4;++t){
      v8f acc = {};
      acc = wmma16(A0, LDV(sW, WT(4+t)), acc);
      acc = wmma16(A1, LDV(sW, WT(8+t)), acc);
      store_tile(acc, lane, [&](int r,int c0,float vv){
        int cg=t*16+c0, k=cg&31;
        sZr[zb + 1*1024 + (cg>>5)*512 + FRAG_LANE(r,k)*16 + FRAG_J(k)] = (_Float16)vv; });
    }
  }
  {
    v16h A0 = silu_frag(LDV(sZr, zb+1024)), A1 = silu_frag(LDV(sZr, zb+1024+512));
#pragma unroll
    for (int t=0;t<4;++t){
      v8f acc = {};
      acc = wmma16(A0, LDV(sW, WT(12+t)), acc);
      acc = wmma16(A1, LDV(sW, WT(16+t)), acc);
      store_tile(acc, lane, [&](int r,int c0,float vv){
        int cg=t*16+c0, k=cg&31;
        sZr[zb + 2*1024 + (cg>>5)*512 + FRAG_LANE(r,k)*16 + FRAG_J(k)] = (_Float16)vv; });
    }
  }
  __syncthreads();
  SWZ_T(0,  M1d, 128, 4, 1, 4, 64);
  SWZ_T(4,  M1c, 64,  1, 2, 4, 64);
  SWZ_T(12, M1b, 64,  1, 2, 4, 64);
  SWZ_T(20, M1a, 64,  1, 2, 1, 8);
  __syncthreads();
  int c = lane;
  for (int mloc=0;mloc<16;++mloc){
    int e = e0+mloc;
    int snd = eidx[e], rcv = eidx[EE+e];
    float gv = scal1[snd*CC+c]*gmsg1[rcv*CC+c];
    sG[gb + FRAG_LANE(mloc,c)*16 + FRAG_J(c)] = (_Float16)gv;  // K=32 chunk 0
  }
  // gh3 = (g_t0 @ Md0^T) * silu'(z3)
  {
    v16h A = LDV(sG, gb);
#pragma unroll
    for (int t=0;t<4;++t){
      v8f acc = {};
      acc = wmma16(A, LDV(sW, WT(0+t)), acc);
      store_tile(acc, lane, [&](int r,int c0,float vv){
        int cg=t*16+c0, k=cg&31;
        float ds = dsiluf((float)sZr[zb + 2*1024 + (cg>>5)*512 + FRAG_LANE(r,k)*16 + FRAG_J(k)]);
        sG[gb + (cg>>5)*512 + FRAG_LANE(r,k)*16 + FRAG_J(k)] = (_Float16)(vv*ds); });
    }
  }
  // gh2
  {
    v16h A0 = LDV(sG, gb), A1 = LDV(sG, gb+512);
#pragma unroll
    for (int t=0;t<4;++t){
      v8f acc = {};
      acc = wmma16(A0, LDV(sW, WT(4+t)), acc);
      acc = wmma16(A1, LDV(sW, WT(8+t)), acc);
      store_tile(acc, lane, [&](int r,int c0,float vv){
        int cg=t*16+c0, k=cg&31;
        float ds = dsiluf((float)sZr[zb + 1*1024 + (cg>>5)*512 + FRAG_LANE(r,k)*16 + FRAG_J(k)]);
        sG[gb + (cg>>5)*512 + FRAG_LANE(r,k)*16 + FRAG_J(k)] = (_Float16)(vv*ds); });
    }
  }
  // gh1
  {
    v16h A0 = LDV(sG, gb), A1 = LDV(sG, gb+512);
#pragma unroll
    for (int t=0;t<4;++t){
      v8f acc = {};
      acc = wmma16(A0, LDV(sW, WT(12+t)), acc);
      acc = wmma16(A1, LDV(sW, WT(16+t)), acc);
      store_tile(acc, lane, [&](int r,int c0,float vv){
        int cg=t*16+c0, k=cg&31;
        float ds = dsiluf((float)sZr[zb + 0*1024 + (cg>>5)*512 + FRAG_LANE(r,k)*16 + FRAG_J(k)]);
        sG[gb + (cg>>5)*512 + FRAG_LANE(r,k)*16 + FRAG_J(k)] = (_Float16)(vv*ds); });
    }
  }
  // grad_rad += gh1 @ Ma^T
  {
    v16h A0 = LDV(sG, gb), A1 = LDV(sG, gb+512);
    v8f acc = {};
    acc = wmma16(A0, LDV(sW, WT(20)), acc);
    acc = wmma16(A1, LDV(sW, WT(21)), acc);
    store_tile(acc, lane, [&](int r,int cc,float vv){
      if (cc < NB) grad_rad[(size_t)(e0+r)*NB + cc] += vv; });
  }
}

// =========================== edge force chain ==========================
__global__ void k_edge_force(const float* __restrict__ u, const float* __restrict__ rr,
                             const float* __restrict__ gea, const float* __restrict__ grad_rad,
                             const int* __restrict__ eidx, float* __restrict__ gpos){
  int e = blockIdx.x*blockDim.x + threadIdx.x;
  if (e >= EE) return;
  float x=u[e*3], y=u[e*3+1], z=u[e*3+2], r=rr[e];
  float invr = 1.f/r;
  float t = r/RMAXf;
  float env=0.f, denv=0.f;
  if (r < RMAXf){
    float t2=t*t, t4=t2*t2, t5=t4*t;
    env  = 1.f - 21.f*t5 + 35.f*t5*t - 15.f*t5*t2;
    denv = (-105.f*t4 + 210.f*t5 - 105.f*t5*t)/RMAXf;
  }
  const float A0=0.63245553f, w=0.62831853f;
  float gr = 0.f;
#pragma unroll
  for (int b=1;b<=NB;++b){
    float arg = w*(float)b*r;
    float sn = __sinf(arg), cs = __cosf(arg);
    float bess  = A0*sn*invr;
    float dbess = A0*(w*(float)b*cs*invr - sn*invr*invr);
    gr += grad_rad[e*NB + b-1]*(dbess*env + bess*denv);
  }
  const float* ge = &gea[e*16];
  const float s3=1.73205081f, s5=2.23606798f, s15=3.87298335f, s105=10.24695077f;
  const float s35_8=2.09165007f, s21_8=1.62018517f, s7=2.64575131f;
  float gx=0.f, gy=0.f, gz=0.f;
  gx += s3*ge[1]; gy += s3*ge[2]; gz += s3*ge[3];
  gx += s15*y*ge[4]; gy += s15*x*ge[4];
  gy += s15*z*ge[5]; gz += s15*y*ge[5];
  gz += 3.f*s5*z*ge[6];
  gx += s15*z*ge[7]; gz += s15*x*ge[7];
  gx += s15*x*ge[8]; gy -= s15*y*ge[8];
  gx += 6.f*s35_8*x*y*ge[9]; gy += 3.f*s35_8*(x*x-y*y)*ge[9];
  gx += s105*y*z*ge[10]; gy += s105*x*z*ge[10]; gz += s105*x*y*ge[10];
  gy += s21_8*(5.f*z*z-1.f)*ge[11]; gz += 10.f*s21_8*y*z*ge[11];
  gz += 0.5f*s7*(15.f*z*z-3.f)*ge[12];
  gx += s21_8*(5.f*z*z-1.f)*ge[13]; gz += 10.f*s21_8*x*z*ge[13];
  gx += s105*x*z*ge[14]; gy -= s105*y*z*ge[14]; gz += 0.5f*s105*(x*x-y*y)*ge[14];
  gx += 3.f*s35_8*(x*x-y*y)*ge[15]; gy -= 6.f*s35_8*x*y*ge[15];
  float gdot = gx*x + gy*y + gz*z;
  float vvx = x*gr + (gx - gdot*x)*invr;
  float vvy = y*gr + (gy - gdot*y)*invr;
  float vvz = z*gr + (gz - gdot*z)*invr;
  int s = eidx[e], d = eidx[EE+e];
  atomicAdd(&gpos[d*3+0],  vvx); atomicAdd(&gpos[d*3+1],  vvy); atomicAdd(&gpos[d*3+2],  vvz);
  atomicAdd(&gpos[s*3+0], -vvx); atomicAdd(&gpos[s*3+1], -vvy); atomicAdd(&gpos[s*3+2], -vvz);
}

__global__ void k_final(const float* __restrict__ gpos, float* __restrict__ out){
  int i = blockIdx.x*blockDim.x + threadIdx.x;
  if (i < NN*3) out[GG + NN + i] = -gpos[i];
}

// =========================== launch ====================================
extern "C" void kernel_launch(void* const* d_in, const int* in_sizes, int n_in,
                              void* d_out, int out_size, void* d_ws, size_t ws_size,
                              hipStream_t stream) {
  const float* pos    = (const float*)d_in[0];
  const float* na     = (const float*)d_in[1];
  const float* shifts = (const float*)d_in[2];
  const int*   eidx   = (const int*)d_in[3];
  const int*   batch  = (const int*)d_in[4];
  const float* aen    = (const float*)d_in[6];
  const float* Wemb   = (const float*)d_in[7];
  const float* Wup1   = (const float*)d_in[8];
  const float* M1a    = (const float*)d_in[9];
  const float* M1b    = (const float*)d_in[10];
  const float* M1c    = (const float*)d_in[11];
  const float* M1d    = (const float*)d_in[12];
  const float* Wl1    = (const float*)d_in[13];
  const float* Wprod1 = (const float*)d_in[14];
  const float* Wup2   = (const float*)d_in[15];
  const float* M2a    = (const float*)d_in[16];
  const float* M2b    = (const float*)d_in[17];
  const float* M2c    = (const float*)d_in[18];
  const float* M2d    = (const float*)d_in[19];
  const float* Wl2    = (const float*)d_in[20];
  const float* Wprod2 = (const float*)d_in[21];
  const float* Wsc2   = (const float*)d_in[22];
  const float* wr1    = (const float*)d_in[23];
  const float* Wr2a   = (const float*)d_in[24];
  const float* wr2b   = (const float*)d_in[25];
  float* out = (float*)d_out;

  char* ws = (char*)d_ws;
  size_t off = 0;
  auto carve = [&](size_t bytes)->void* {
    void* p = ws + off;
    off = (off + bytes + 255) & ~(size_t)255;
    return p;
  };
  float*     u        = (float*)carve((size_t)3*EE*4);
  float*     rr       = (float*)carve((size_t)EE*4);
  float*     sph      = (float*)carve((size_t)16*EE*4);
  float*     rad      = (float*)carve((size_t)NB*EE*4);
  _Float16*  tw2      = (_Float16*)carve((size_t)128*EE*2);
  int*       nodek    = (int*)carve((size_t)NN*4);
  float*     scal1    = (float*)carve((size_t)CC*NN*4);
  float*     msg1     = (float*)carve((size_t)CC*NN*4);
  float*     A1_0     = (float*)carve((size_t)CC*NN*4);
  float*     nf1_0    = (float*)carve((size_t)CC*NN*4);
  float*     scal2    = (float*)carve((size_t)CC*NN*4);
  float*     msg2     = (float*)carve((size_t)512*NN*4);
  float*     A2       = (float*)carve((size_t)512*NN*4);
  float*     zbuf     = (float*)carve((size_t)16*NN*4);
  float*     gmsg2    = (float*)carve((size_t)512*NN*4);
  float*     gnf1     = (float*)carve((size_t)CC*NN*4);
  float*     gscal2   = (float*)carve((size_t)CC*NN*4);
  float*     gmsg1    = (float*)carve((size_t)CC*NN*4);
  float*     grad_rad = (float*)carve((size_t)NB*EE*4);
  float*     gea      = (float*)carve((size_t)16*EE*4);
  float*     gpos     = (float*)carve((size_t)3*NN*4);

  k_zero<<<(CC*NN+255)/256, 256, 0, stream>>>(msg1,   CC*NN);
  k_zero<<<(512*NN+255)/256,256, 0, stream>>>(msg2,   512*NN);
  k_zero<<<(CC*NN+255)/256, 256, 0, stream>>>(gscal2, CC*NN);
  k_zero<<<(3*NN+255)/256,  256, 0, stream>>>(gpos,   3*NN);
  k_zero<<<1, 32, 0, stream>>>(out, GG);

  k_edge_geom<<<(EE+255)/256, 256, 0, stream>>>(pos, shifts, eidx, u, rr, sph, rad);
  k_node_embed<<<NN, 32, 0, stream>>>(na, Wemb, Wup1, nodek, scal1);
  k_mlp1_fwd<<<EE/32, 64, 0, stream>>>(rad, M1a, M1b, M1c, M1d, eidx, scal1, msg1);
  k_node_fwd1<<<NN, 32, 0, stream>>>(msg1, Wl1, Wprod1, Wup2, nodek, A1_0, nf1_0, scal2);
  k_mlp2_fwd<<<EE/32, 64, 0, stream>>>(rad, M2a, M2b, M2c, M2d, eidx, scal2, sph, tw2, msg2);
  k_node_fwd2<<<NN, 32, 0, stream>>>(msg2, Wl2, Wprod2, Wsc2, nodek, nf1_0,
                                     aen, wr1, Wr2a, wr2b, batch, A2, zbuf, out);
  k_node_bwd1<<<NN, 32, 0, stream>>>(zbuf, wr2b, Wr2a, Wsc2, Wprod2, Wl2, nodek, A2, wr1,
                                     gnf1, gmsg2);
  k_mlp2_bwd<<<EE/32, 64, 0, stream>>>(rad, M2a, M2b, M2c, M2d, eidx, scal2, sph, tw2, gmsg2,
                                       gscal2, gea, grad_rad);
  k_node_bwd2<<<NN, 32, 0, stream>>>(gscal2, Wup2, gnf1, Wprod1, nodek, A1_0, Wl1, gmsg1);
  k_mlp1_bwd<<<EE/32, 64, 0, stream>>>(rad, M1a, M1b, M1c, M1d, eidx, scal1, gmsg1, grad_rad);
  k_edge_force<<<(EE+255)/256, 256, 0, stream>>>(u, rr, gea, grad_rad, eidx, gpos);
  k_final<<<(3*NN+255)/256, 256, 0, stream>>>(gpos, out);
}
